// GRUModel_45818711114286
// MI455X (gfx1250) — matrix operand order repaired and checked by reading, not compiled, not run on hardware
//
#include <hip/hip_runtime.h>
#include <hip/hip_bf16.h>

// ---------------------------------------------------------------------------
// GRU (5 layers, PyTorch gate conventions) for MI455X / gfx1250 (wave32, WMMA)
//   phase 0: one-shot f32 -> bf16 conversion of x and all weights (packed cvt)
//   phase A: gx = in @ W_ih^T + b_ih      -> v_wmma_f32_16x16x32_bf16 tiles
//   phase B: sequential scan; W_hh, h, and async-staged gx slice in LDS;
//            WMMA per timestep; gx(t+1) via GLOBAL_LOAD_ASYNC_TO_LDS_B128
//   phase C: final linear on last timestep
// ---------------------------------------------------------------------------

#define B_       64
#define T_       512
#define IN_      512
#define H_       128
#define G3_      384      // 3*H
#define OUT_     96
#define NL_      5

typedef __bf16 bf16x16 __attribute__((ext_vector_type(16)));
typedef __bf16 bf16x8  __attribute__((ext_vector_type(8)));
typedef float  f32x8   __attribute__((ext_vector_type(8)));
typedef int    v4i     __attribute__((ext_vector_type(4)));

#if defined(__has_builtin)
#  if __has_builtin(__builtin_amdgcn_global_load_async_to_lds_b128)
#    define HAVE_ASYNC_LDS 1
#  endif
#endif
#ifndef HAVE_ASYNC_LDS
#  define HAVE_ASYNC_LDS 0
#endif

// 16-byte global -> LDS copy; async DMA when the gfx1250 builtin is available.
static __device__ __forceinline__ void copy16_to_lds(const float* g, float* l) {
#if HAVE_ASYNC_LDS
    __builtin_amdgcn_global_load_async_to_lds_b128(
        (__attribute__((address_space(1))) v4i*)(void*)g,
        (__attribute__((address_space(3))) v4i*)(void*)l, 0, 0);
#else
    *(float4*)l = *(const float4*)g;     // synchronous staged fallback
#endif
}
static __device__ __forceinline__ void async_drain() {
#if HAVE_ASYNC_LDS
#  if __has_builtin(__builtin_amdgcn_s_wait_asynccnt)
    __builtin_amdgcn_s_wait_asynccnt(0);
#  else
    asm volatile("s_wait_asynccnt 0x0" ::: "memory");
#  endif
#endif
}

static __device__ __forceinline__ float fast_rcp(float x) {
    return __builtin_amdgcn_rcpf(x);
}
static __device__ __forceinline__ float sigmoidf_(float x) {
    return fast_rcp(1.0f + __expf(-x));
}
static __device__ __forceinline__ float tanhf_(float x) {
    float e = __expf(2.0f * x);
    return 1.0f - 2.0f * fast_rcp(e + 1.0f);
}

// ---------------------------------------------------------------------------
// elementwise f32 -> bf16 (native cast -> v_cvt packed), 8 elements / thread
// ---------------------------------------------------------------------------
__global__ void __launch_bounds__(256)
cvt_bf16(const float* __restrict__ src, __bf16* __restrict__ dst, int n) {
    const int i = (blockIdx.x * blockDim.x + threadIdx.x) * 8;
    if (i >= n) return;
    bf16x8 v;
    #pragma unroll
    for (int j = 0; j < 8; ++j) v[j] = (__bf16)src[i + j];
    *(bf16x8*)(dst + i) = v;
}

// ---------------------------------------------------------------------------
// gemm_gates: out[M,384] = in[M,K](bf16) @ w[384,K](bf16)^T + bias[384](f32)
// one wave == one 16x16 output tile; block = 128 threads = 4 waves
// grid = (M/16, 384/16/4)
// ---------------------------------------------------------------------------
__global__ void __launch_bounds__(128)
gemm_gates(const __bf16* __restrict__ in, const __bf16* __restrict__ w,
           const float* __restrict__ bias, float* __restrict__ out, int K) {
    const int lane = threadIdx.x & 31;
    const int wv   = threadIdx.x >> 5;
    const int m0   = blockIdx.x * 16;
    const int n0   = (blockIdx.y * 4 + wv) * 16;

    const int arow = m0 + (lane & 15);
    const int ncol = n0 + (lane & 15);
    const int khA  = (lane >> 4) * 16;     // A K-half select
    const int khB  = (lane >> 4) * 16;     // B K-half select

    f32x8 c = {};
    for (int k0 = 0; k0 < K; k0 += 32) {
        const __bf16* ap = in + (size_t)arow * K + k0 + khA;
        bf16x8 a0 = *(const bf16x8*)(ap);
        bf16x8 a1 = *(const bf16x8*)(ap + 8);
        bf16x16 a = __builtin_shufflevector(a0, a1, 0, 1, 2, 3, 4, 5, 6, 7,
                                                    8, 9, 10, 11, 12, 13, 14, 15);
        bf16x16 b = *(const bf16x16*)(w + (size_t)ncol * K + k0 + khB);
        c = __builtin_amdgcn_wmma_f32_16x16x32_bf16(false, a, false, b,
                                                    (short)0, c, false, false);
    }
    const float bv  = bias[ncol];
    const int mrow  = m0 + ((lane >> 4) * 8);
    #pragma unroll
    for (int r = 0; r < 8; ++r)
        out[(size_t)(mrow + r) * G3_ + ncol] = c[r] + bv;
}

// ---------------------------------------------------------------------------
// gru_scan: one layer. Single persistent workgroup (1024 thr = 32 waves).
// LDS: w_hh bf16 [384][128] | h bf16 [64][128] | gh f32 [64][384] | gx stage [64][384]
//      = 98304 + 16384 + 98304 + 98304 = 311296 B  (<= 320KB WGP LDS)
// Per timestep:
//   WMMA phase (lds_hb x lds_w -> lds_gh), drain own async loads, barrier,
//   gate phase (pure LDS reads), barrier, issue async loads of gx(t+1).
// h f32 state lives in 8 registers per thread (fixed element ownership).
// ---------------------------------------------------------------------------
__global__ void __launch_bounds__(1024)
gru_scan(const float* __restrict__ gx, const __bf16* __restrict__ whh,
         const float* __restrict__ bhh, __bf16* __restrict__ hseq_bf) {
    extern __shared__ char smem[];
    __bf16* lds_w  = (__bf16*)smem;                                   // 98304 B
    __bf16* lds_hb = (__bf16*)(smem + G3_ * H_ * 2);                  // 16384 B
    float*  lds_gh = (float*)(smem + G3_ * H_ * 2 + B_ * H_ * 2);     // 98304 B
    float*  lds_gx = (float*)((char*)lds_gh + B_ * G3_ * 4);          // 98304 B

    const int tid = threadIdx.x;

    // stage bf16 recurrent weights into LDS (wide copies), zero bf16 state
    for (int i = tid * 8; i < G3_ * H_; i += 1024 * 8)
        *(bf16x8*)(lds_w + i) = *(const bf16x8*)(whh + i);
    {
        bf16x8 z8 = {};
        *(bf16x8*)(lds_hb + tid * 8) = z8;          // 8192 bf16 = 1024*8
    }

    // per-thread slice of the gx staging copy: 6144 16B vectors over 1024 thr
    size_t goff[6];    // element offset of vector i at t=0 (add t*G3_ per step)
    int    loff[6];    // float offset in lds_gx
    #pragma unroll
    for (int i = 0; i < 6; ++i) {
        const int v  = i * 1024 + tid;        // 0..6143
        const int m  = v / 96;                // batch row (96 vec16 per row)
        const int c4 = v % 96;                // 16B chunk within the 384 cols
        goff[i] = (size_t)m * T_ * G3_ + c4 * 4;
        loff[i] = v * 4;
    }
    // preload gx(t=0)
    #pragma unroll
    for (int i = 0; i < 6; ++i) copy16_to_lds(gx + goff[i], lds_gx + loff[i]);

    __syncthreads();   // weights, hb=0 visible (async gx drained before use below)

    const int lane = tid & 31;
    const int wv   = tid >> 5;          // 0..31
    const int m0   = (wv & 3) * 16;     // 4 m-tiles cover B=64
    const int ng   = wv >> 2;           // 8 groups x 3 n-tiles cover 384
    const int arow = m0 + (lane & 15);
    const int khA  = (lane >> 4) * 16;
    const int khB  = (lane >> 4) * 16;
    const int nlan = lane & 15;
    const int mrow = m0 + ((lane >> 4) * 8);

    float hreg[8];
    #pragma unroll
    for (int e = 0; e < 8; ++e) hreg[e] = 0.0f;

    for (int t = 0; t < T_; ++t) {
        // ---- gh = h @ W_hh^T (bf16 WMMA, f32 accumulate) --------------------
        #pragma unroll
        for (int nn = 0; nn < 3; ++nn) {
            const int n0 = (ng * 3 + nn) * 16;
            f32x8 c = {};
            #pragma unroll
            for (int kk = 0; kk < 4; ++kk) {
                const int k0 = kk * 32;
                const __bf16* ap = lds_hb + arow * H_ + k0 + khA;
                bf16x8 a0 = *(const bf16x8*)(ap);
                bf16x8 a1 = *(const bf16x8*)(ap + 8);
                bf16x16 a = __builtin_shufflevector(a0, a1, 0, 1, 2, 3, 4, 5, 6, 7,
                                                            8, 9, 10, 11, 12, 13, 14, 15);
                bf16x16 b = *(const bf16x16*)(lds_w + (n0 + nlan) * H_ + k0 + khB);
                c = __builtin_amdgcn_wmma_f32_16x16x32_bf16(false, a, false, b,
                                                            (short)0, c, false, false);
            }
            #pragma unroll
            for (int r = 0; r < 8; ++r)
                lds_gh[(mrow + r) * G3_ + n0 + nlan] = c[r];
        }
        async_drain();       // this wave's gx(t) loads complete
        __syncthreads();     // all waves' gx(t) + gh(t) visible

        // ---- gates + state update (64*128 elems, 8/thread, LDS-only reads) --
        #pragma unroll
        for (int e = 0; e < 8; ++e) {
            const int idx = e * 1024 + tid;        // coalesced
            const int m = idx >> 7;                // batch row
            const int j = idx & (H_ - 1);          // hidden col
            const float xr = lds_gx[m * G3_ + j];
            const float xz = lds_gx[m * G3_ + H_ + j];
            const float xn = lds_gx[m * G3_ + 2 * H_ + j];
            const float hr = lds_gh[m * G3_ + j]          + bhh[j];
            const float hz = lds_gh[m * G3_ + H_ + j]     + bhh[H_ + j];
            const float hn = lds_gh[m * G3_ + 2 * H_ + j] + bhh[2 * H_ + j];
            const float r  = sigmoidf_(xr + hr);
            const float z  = sigmoidf_(xz + hz);
            const float n  = tanhf_(xn + r * hn);
            const float hv = (1.0f - z) * n + z * hreg[e];
            hreg[e] = hv;
            const __bf16 hb = (__bf16)hv;
            lds_hb[m * H_ + j] = hb;
            hseq_bf[((size_t)m * T_ + t) * H_ + j] = hb;
        }
        __syncthreads();     // gx(t) fully consumed; hb(t) ready for next WMMA

        // ---- kick off async staging of gx(t+1), hidden behind next WMMA ----
        if (t + 1 < T_) {
            const size_t tb = (size_t)(t + 1) * G3_;
            #pragma unroll
            for (int i = 0; i < 6; ++i)
                copy16_to_lds(gx + tb + goff[i], lds_gx + loff[i]);
        }
    }
}

// ---------------------------------------------------------------------------
// final linear on last timestep: out[b,o] = h_last[b,:] . fc_w[o,:] + fc_b[o]
// ---------------------------------------------------------------------------
__global__ void fc_kernel(const __bf16* __restrict__ hseq_bf, const float* __restrict__ fw,
                          const float* __restrict__ fb, float* __restrict__ out) {
    const int idx = blockIdx.x * blockDim.x + threadIdx.x;
    if (idx >= B_ * OUT_) return;
    const int b = idx / OUT_;
    const int o = idx % OUT_;
    const __bf16* h = hseq_bf + ((size_t)b * T_ + (T_ - 1)) * H_;
    const float*  w = fw + (size_t)o * H_;
    float acc = fb[o];
    #pragma unroll 8
    for (int j = 0; j < H_; ++j) acc += (float)h[j] * w[j];
    out[idx] = acc;
}

// ---------------------------------------------------------------------------
extern "C" void kernel_launch(void* const* d_in, const int* in_sizes, int n_in,
                              void* d_out, int out_size, void* d_ws, size_t ws_size,
                              hipStream_t stream) {
    (void)in_sizes; (void)n_in; (void)out_size; (void)ws_size;
    const float* x     = (const float*)d_in[0];
    const float* w_ih0 = (const float*)d_in[1];
    const float* w_ihr = (const float*)d_in[2];
    const float* w_hh  = (const float*)d_in[3];
    const float* b_ih  = (const float*)d_in[4];
    const float* b_hh  = (const float*)d_in[5];
    const float* fc_w  = (const float*)d_in[6];
    const float* fc_b  = (const float*)d_in[7];
    float* out = (float*)d_out;

    // ---- workspace carve (bytes) ----
    char* ws = (char*)d_ws;
    float*  gx    = (float*)ws;                              // 64*512*384 f32 = 50331648 B
    __bf16* xbf   = (__bf16*)(ws + 50331648);                // 64*512*512 bf16 = 33554432 B
    __bf16* hbfA  = (__bf16*)(ws + 83886080);                // 64*512*128 bf16 =  8388608 B
    __bf16* hbfB  = (__bf16*)(ws + 92274688);                //                  8388608 B
    __bf16* wih0b = (__bf16*)(ws + 100663296);               // 384*512 bf16   =   393216 B
    __bf16* wihrb = (__bf16*)(ws + 101056512);               // 4*384*128 bf16 =   393216 B
    __bf16* whhb  = (__bf16*)(ws + 101449728);               // 5*384*128 bf16 =   491520 B

    // ---- phase 0: one-shot bf16 conversions ----
    {
        const int nx = B_ * T_ * IN_;                 // 16M
        cvt_bf16<<<(nx / 8 + 255) / 256, 256, 0, stream>>>(x, xbf, nx);
        const int n0 = G3_ * IN_;                     // w_ih0
        cvt_bf16<<<(n0 / 8 + 255) / 256, 256, 0, stream>>>(w_ih0, wih0b, n0);
        const int nr = (NL_ - 1) * G3_ * H_;          // w_ih_rest
        cvt_bf16<<<(nr / 8 + 255) / 256, 256, 0, stream>>>(w_ihr, wihrb, nr);
        const int nh = NL_ * G3_ * H_;                // w_hh
        cvt_bf16<<<(nh / 8 + 255) / 256, 256, 0, stream>>>(w_hh, whhb, nh);
    }

    const int M = B_ * T_;
    const dim3 gblk(128);
    const dim3 ggrd(M / 16, G3_ / 16 / 4);
    const size_t scan_lds = (size_t)G3_ * H_ * 2   // w_hh bf16
                          + (size_t)B_ * H_ * 2    // h bf16
                          + (size_t)B_ * G3_ * 4   // gh f32
                          + (size_t)B_ * G3_ * 4;  // gx stage f32 -> 311296 B

    // ---- layer 0 (K = 512) ----
    gemm_gates<<<ggrd, gblk, 0, stream>>>(xbf, wih0b, b_ih, gx, IN_);
    gru_scan<<<1, 1024, scan_lds, stream>>>(gx, whhb, b_hh, hbfA);

    __bf16* cur = hbfA;
    __bf16* nxt = hbfB;
    for (int l = 1; l < NL_; ++l) {
        gemm_gates<<<ggrd, gblk, 0, stream>>>(
            cur, wihrb + (size_t)(l - 1) * G3_ * H_, b_ih + (size_t)l * G3_, gx, H_);
        gru_scan<<<1, 1024, scan_lds, stream>>>(
            gx, whhb + (size_t)l * G3_ * H_, b_hh + (size_t)l * G3_, nxt);
        __bf16* tmp = cur; cur = nxt; nxt = tmp;
    }

    fc_kernel<<<(B_ * OUT_ + 255) / 256, 256, 0, stream>>>(cur, fc_w, fc_b, out);
}
